// LstmForClassification_27805618274778
// MI455X (gfx1250) — matrix operand order repaired and checked by reading, not compiled, not run on hardware
//
#include <hip/hip_runtime.h>
#include <hip/hip_bf16.h>

// ---------------------------------------------------------------------------
// LSTM-for-classification on gfx1250 (MI455X).
// bf16 WMMA (v_wmma_f32_16x16x32_bf16) with f32 accumulation; fp32 cell state.
// One persistent kernel per layer (grid-wide barrier between timesteps);
// A-tiles staged to LDS via async global->LDS (ASYNCcnt) when available.
// ---------------------------------------------------------------------------

typedef __bf16 bf16_t;
typedef __attribute__((ext_vector_type(16))) __bf16 v16bf;
typedef __attribute__((ext_vector_type(8)))  float  v8f;

struct alignas(16) U128 { unsigned int x, y, z, w; };
struct U256 { U128 lo, hi; };

// pointer types for the async global->LDS builtin (probe: arg0 is non-const
// int4(vector_size 16) addrspace(1)*; arg1 is the LDS side)
typedef int v4i_t __attribute__((vector_size(16)));
typedef __attribute__((address_space(1))) v4i_t* gp128_t;
typedef __attribute__((address_space(3))) v4i_t* lp128_t;

constexpr int kB  = 32;      // batch
constexpr int kT  = 512;     // time steps
constexpr int kH  = 1024;    // hidden
constexpr int kNL = 5;       // labels
constexpr int kK  = 2 * kH;  // fused K = [x ; h] = 2048
constexpr int KC  = 512;     // K-chunk staged in LDS
constexpr int ASTR = KC + 8; // padded LDS row stride (elems): 1040B -> 4-bank skew
constexpr int kNB = 16;      // persistent grid size (co-resident)

#if __has_builtin(__builtin_amdgcn_global_load_async_to_lds_b128)
#define ASYNC_LDS 1
#else
#define ASYNC_LDS 0
#endif

__device__ __forceinline__ float sigf(float x) {
  return 1.0f / (1.0f + __expf(-x));
}
__device__ __forceinline__ float tanhf_fast(float x) {
  return 2.0f * sigf(2.0f * x) - 1.0f;
}

// --- convert W_ih/W_hh (fp32) -> concatenated bf16 W_cat[layer][4H][2H] -----
__global__ __launch_bounds__(256) void convw_kernel(
    const float* __restrict__ Wih, const float* __restrict__ Whh,
    bf16_t* __restrict__ Wb) {
  size_t gid = (size_t)blockIdx.x * 256 + threadIdx.x;   // < L*4H*H = 2^23
  size_t k = gid & (kH - 1);
  size_t n = (gid >> 10) & 4095;
  size_t l = gid >> 22;
  bf16_t* wrow = Wb + ((l * (size_t)4096 + n) * kK);
  wrow[k]      = (bf16_t)Wih[gid];
  wrow[kH + k] = (bf16_t)Whh[gid];
}

// --- embedding gather -> bf16 time-major X0[T,B,H] --------------------------
__global__ __launch_bounds__(256) void embed_kernel(
    const int* __restrict__ x, const float* __restrict__ emb,
    bf16_t* __restrict__ X0) {
  int tb = blockIdx.x;               // t*B + b
  int t = tb >> 5, b = tb & 31;
  int idx = x[b * kT + t];
  const float* src = emb + (size_t)idx * kH;
  bf16_t* dst = X0 + (size_t)tb * kH;
  for (int i = threadIdx.x; i < kH; i += 256) dst[i] = (bf16_t)src[i];
}

// --- zero recurrent state + grid-barrier counter -----------------------------
__global__ __launch_bounds__(256) void zero_state_kernel(
    float* __restrict__ Cst, bf16_t* __restrict__ Hb,
    unsigned* __restrict__ cnt) {
  int i = blockIdx.x * 256 + threadIdx.x;   // < 32768
  Cst[i] = 0.0f;
  Hb[i]  = (bf16_t)0.0f;
  if (i == 0) *cnt = 0u;
}

// --- device-wide barrier (monotonic counter; 16 co-resident blocks) ----------
__device__ __forceinline__ void grid_barrier(unsigned* cnt, unsigned target) {
  __builtin_amdgcn_fence(__ATOMIC_RELEASE, "agent");
  __syncthreads();
  if (threadIdx.x == 0) {
    __hip_atomic_fetch_add(cnt, 1u, __ATOMIC_RELAXED, __HIP_MEMORY_SCOPE_AGENT);
    while (__hip_atomic_load(cnt, __ATOMIC_RELAXED, __HIP_MEMORY_SCOPE_AGENT) <
           target)
      __builtin_amdgcn_s_sleep(1);
  }
  __syncthreads();
  __builtin_amdgcn_fence(__ATOMIC_ACQUIRE, "agent");
}

// --- persistent per-layer LSTM kernel ---------------------------------------
// 16 blocks x 256 threads (8 waves). Block owns 64 hidden units; wave w:
// m-tile = (w&1)*16, hidden sub-tile = (w>>1)*16 -> owns all 4 gate tiles of
// its 16 hidden units, so the sigmoid/tanh cell update is wave-local.
// Per step: gates[32,4096] = [x_t ; h_{t-1}][32,2048] @ W_cat^T via bf16 WMMA.
__global__ __launch_bounds__(256) void lstm_layer_kernel(
    const bf16_t* __restrict__ Xin,   // [T*B*H] time-major
    const bf16_t* __restrict__ Wcat,  // [4H * 2H] bf16 (row n: [Wih_n|Whh_n])
    const float*  __restrict__ bias,  // [4H]
    float*        __restrict__ Cst,   // [B*H] fp32 cell state
    bf16_t*       __restrict__ Hb0,   // [B*H] h ping (zeroed)
    bf16_t*       __restrict__ Hb1,   // [B*H] h pong
    float*        __restrict__ Hfp,   // [B*H] fp32 h_T (for head)
    bf16_t*       __restrict__ Xout,  // [T*B*H] layer output or nullptr
    unsigned*     __restrict__ cnt) { // grid barrier counter (zeroed)
  __shared__ __align__(16) bf16_t sA[kB * ASTR];  // 33,280 B

  const int tid   = threadIdx.x;
  const int lane  = tid & 31;
  const int l15   = lane & 15;
  const int hi    = lane >> 4;            // half-wave select
  const int wave  = tid >> 5;
  const int mtile = (wave & 1) << 4;      // 0 | 16
  const int nsub  = wave >> 1;            // 0..3
  const int hbase = blockIdx.x * 64 + nsub * 16;

  // B operand: one weight row (= gate output unit) per lane column.
  const bf16_t* wcol = Wcat + (size_t)(hbase + l15) * kK;
  const int arow = mtile + l15;
  const int col  = hbase + l15;
  const float bi = bias[col];
  const float bf = bias[col + kH];
  const float bg = bias[col + 2 * kH];
  const float bo = bias[col + 3 * kH];

  for (int t = 0; t < kT; ++t) {
    if (t) grid_barrier(cnt, (unsigned)t * kNB);

    const bf16_t* xrow = Xin + (size_t)t * kB * kH;
    const bf16_t* Hin  = (t & 1) ? Hb1 : Hb0;
    bf16_t*       Hout = (t & 1) ? Hb0 : Hb1;

    v8f acc[4] = {v8f{}, v8f{}, v8f{}, v8f{}};  // i, f, g, o

    for (int kc = 0; kc < kK / KC; ++kc) {
      // ---- stage A chunk [32 x 512] bf16 into LDS --------------------------
      {
        const bf16_t* src = (kc < 2) ? (xrow + kc * KC) : (Hin + (kc - 2) * KC);
#pragma unroll
        for (int i = 0; i < 8; ++i) {
          int c   = tid + i * 256;          // 0..2047 16B-chunks
          int row = c >> 6;
          int k8  = (c & 63) << 3;
#if ASYNC_LDS
          __builtin_amdgcn_global_load_async_to_lds_b128(
              (gp128_t)(src + (size_t)row * kH + k8),
              (lp128_t)(&sA[row * ASTR + k8]), 0, 0);
#else
          *(U128*)(&sA[row * ASTR + k8]) =
              *(const U128*)(src + (size_t)row * kH + k8);
#endif
        }
      }
      // warm next K-chunk of L2-resident weights
      if (kc + 1 < kK / KC) {
#pragma unroll
        for (int g = 0; g < 4; ++g)
          __builtin_prefetch(wcol + (size_t)g * kH * kK + (kc + 1) * KC, 0, 3);
      }
#if ASYNC_LDS
#if __has_builtin(__builtin_amdgcn_s_wait_asynccnt)
      __builtin_amdgcn_s_wait_asynccnt(0);
#else
      asm volatile("s_wait_asynccnt 0" ::: "memory");
#endif
#endif
      __syncthreads();

      // ---- K inner loop: 16 steps of 32, 4 WMMAs each ----------------------
#pragma unroll 2
      for (int ks = 0; ks < KC; ks += 32) {
        // A fragment 16x32 (lanes 0-15: K 0..15; lanes 16-31: K 16..31)
        U256 ap;
        const int ka = ks + (hi << 4);
        ap.lo = *(const U128*)(&sA[arow * ASTR + ka]);
        ap.hi = *(const U128*)(&sA[arow * ASTR + ka + 8]);
        const v16bf a = __builtin_bit_cast(v16bf, ap);

        // B fragment 32x16 (lanes 0-15: K 0..15; lanes 16-31: K 16..31)
        const int kg = kc * KC + ks + (hi << 4);
#pragma unroll
        for (int g = 0; g < 4; ++g) {
          const bf16_t* wp = wcol + (size_t)g * kH * kK + kg;
          U256 bp;
          bp.lo = *(const U128*)(wp);
          bp.hi = *(const U128*)(wp + 8);
          acc[g] = __builtin_amdgcn_wmma_f32_16x16x32_bf16(
              false, a, false, __builtin_bit_cast(v16bf, bp),
              (short)0, acc[g], false, false);
        }
      }
      __syncthreads();
    }

    // ---- fused LSTM cell update (C/D layout: vgpr j -> M=j(+8), N=l15) -----
#pragma unroll
    for (int j = 0; j < 8; ++j) {
      const int m = mtile + j + (hi << 3);
      const size_t idx = (size_t)m * kH + col;
      const float gi = acc[0][j] + bi;
      const float gf = acc[1][j] + bf;
      const float gg = acc[2][j] + bg;
      const float go = acc[3][j] + bo;
      const float cn = sigf(gf) * Cst[idx] + sigf(gi) * tanhf_fast(gg);
      Cst[idx] = cn;
      const float h = sigf(go) * tanhf_fast(cn);
      const bf16_t hb = (bf16_t)h;
      Hout[idx] = hb;
      if (Xout) Xout[(size_t)t * kB * kH + idx] = hb;
      if (t == kT - 1) Hfp[idx] = h;
    }
  }
}

// --- classifier head: logits, log-softmax, NLL mean --------------------------
__global__ __launch_bounds__(256) void head_kernel(
    const float* __restrict__ Hfp, const float* __restrict__ fcW,
    const float* __restrict__ fcb, const int* __restrict__ labels,
    float* __restrict__ out) {
  __shared__ float slog[kB * kNL];
  __shared__ float sloss[kB];
  const int tid = threadIdx.x;
  if (tid < kB * kNL) {
    const int m = tid / kNL, c = tid % kNL;
    const float* hv = Hfp + (size_t)m * kH;
    const float* wv = fcW + (size_t)c * kH;
    float s = fcb[c];
    for (int k = 0; k < kH; ++k) s = fmaf(hv[k], wv[k], s);
    slog[tid] = s;
    out[1 + tid] = s;           // logits, row-major [B, NL]
  }
  __syncthreads();
  if (tid < kB) {
    float mx = -1e30f;
    for (int c = 0; c < kNL; ++c) mx = fmaxf(mx, slog[tid * kNL + c]);
    float sum = 0.0f;
    for (int c = 0; c < kNL; ++c) sum += __expf(slog[tid * kNL + c] - mx);
    const int lab = labels[tid];
    sloss[tid] = -(slog[tid * kNL + lab] - mx - __logf(sum));
  }
  __syncthreads();
  if (tid == 0) {
    float s = 0.0f;
    for (int m = 0; m < kB; ++m) s += sloss[m];
    out[0] = s / (float)kB;     // loss
  }
}

// ---------------------------------------------------------------------------
extern "C" void kernel_launch(void* const* d_in, const int* in_sizes, int n_in,
                              void* d_out, int out_size, void* d_ws,
                              size_t ws_size, hipStream_t stream) {
  (void)in_sizes; (void)n_in; (void)out_size; (void)ws_size;
  const int*   x      = (const int*)d_in[0];
  const int*   labels = (const int*)d_in[1];
  const float* emb    = (const float*)d_in[2];
  const float* Wih    = (const float*)d_in[3];
  const float* Whh    = (const float*)d_in[4];
  const float* bias   = (const float*)d_in[5];
  const float* fcW    = (const float*)d_in[6];
  const float* fcb    = (const float*)d_in[7];
  float* out = (float*)d_out;

  // workspace carve-out (~101 MB)
  char* ws = (char*)d_ws;
  bf16_t* Wb  = (bf16_t*)ws; ws += (size_t)2 * 4096 * kK * sizeof(bf16_t);
  bf16_t* X0  = (bf16_t*)ws; ws += (size_t)kT * kB * kH * sizeof(bf16_t);
  bf16_t* X1  = (bf16_t*)ws; ws += (size_t)kT * kB * kH * sizeof(bf16_t);
  bf16_t* Hb0 = (bf16_t*)ws; ws += (size_t)kB * kH * sizeof(bf16_t);
  bf16_t* Hb1 = (bf16_t*)ws; ws += (size_t)kB * kH * sizeof(bf16_t);
  float*  Cst = (float*)ws;  ws += (size_t)kB * kH * sizeof(float);
  float*  Hfp = (float*)ws;  ws += (size_t)kB * kH * sizeof(float);
  unsigned* cnt = (unsigned*)ws; ws += 256;

  convw_kernel<<<32768, 256, 0, stream>>>(Wih, Whh, Wb);
  embed_kernel<<<kT * kB, 256, 0, stream>>>(x, emb, X0);

  for (int l = 0; l < 2; ++l) {
    zero_state_kernel<<<(kB * kH) / 256, 256, 0, stream>>>(Cst, Hb0, cnt);
    const bf16_t* Xin  = l ? X1 : X0;
    bf16_t*       Xout = l ? (bf16_t*)nullptr : X1;  // layer-2 output unused
    const bf16_t* Wl   = Wb + (size_t)l * 4096 * kK;
    const float*  bl   = bias + l * 4096;
    lstm_layer_kernel<<<kNB, 256, 0, stream>>>(Xin, Wl, bl, Cst, Hb0, Hb1,
                                               Hfp, Xout, cnt);
  }

  head_kernel<<<1, 256, 0, stream>>>(Hfp, fcW, fcb, labels, out);
}
